// EncoderLayer_60936995996019
// MI455X (gfx1250) — compile-verified, matrix operand order checked by reading, not run on hardware
//
#include <hip/hip_runtime.h>
#include <hip/hip_bf16.h>

#define B_  2
#define S_  2048
#define D_  1024
#define H_  16
#define DH_ 64
#define FF_ 4096
#define M_  (B_ * S_)   // 4096 token rows

#define USE_TDM 1

typedef __attribute__((ext_vector_type(16))) __bf16 v16bf;
typedef __attribute__((ext_vector_type(8)))  float  v8f;
typedef __attribute__((ext_vector_type(4)))  unsigned int u32x4;
typedef __attribute__((ext_vector_type(8)))  int i32x8;
typedef __attribute__((ext_vector_type(4)))  int i32x4;

// A/B fragment for v_wmma_f32_16x16x32_bf16:
// lane (l16 = lane&15, half = lane>>4) holds row/col l16, with two contiguous
// 8-element K-chunks at K = half*8 and K = 16 + half*8 (ISA 7.12.2).
union FragU { struct { uint4 lo, hi; } q; v16bf v; };

__device__ __forceinline__ v16bf load_frag(const __bf16* base) {
    FragU f;
    f.q.lo = *(const uint4*)(base);
    f.q.hi = *(const uint4*)(base + 16);
    return f.v;
}

// ---------------------------------------------------------------------------
// TDM: async 2D bf16 tile load (tile_y rows x tile_x elems, row stride in
// elements) from global to LDS (packed row-major). Issue once per workgroup
// (from one wave); complete with s_wait_tensorcnt(0) + barrier.
// D# layout per CDNA5 ISA ch.8 (group0 128b, group1 256b; groups 2/3 zero).
// ---------------------------------------------------------------------------
__device__ __forceinline__ void tdm_load_2d_bf16(void* lds_dst, const void* gsrc,
                                                 unsigned tile_x, unsigned tile_y,
                                                 unsigned long long stride_elems) {
    unsigned long long ga = (unsigned long long)(uintptr_t)gsrc;   // byte addr
    unsigned lds_addr = (unsigned)(uintptr_t)lds_dst;              // LDS byte addr

    u32x4 g0 = {};
    g0[0] = 1u;                                            // count=1 (user D#)
    g0[1] = lds_addr;                                      // lds_addr [63:32]
    g0[2] = (unsigned)(ga & 0xFFFFFFFFu);                  // global_addr lo
    g0[3] = (unsigned)((ga >> 32) & 0x01FFFFFFu)           // global_addr hi (57b)
          | (2u << 30);                                    // type = 2 ("image")

    i32x8 g1 = {};
    g1[0] = (int)(1u << 16);                               // data_size=1 (2B), mask=0
    g1[1] = (int)((tile_x & 0xFFFFu) << 16);               // tensor_dim0[15:0]
    g1[2] = (int)(((tile_x >> 16) & 0xFFFFu)               // tensor_dim0[31:16]
          | ((tile_y & 0xFFFFu) << 16));                   // tensor_dim1[15:0]
    g1[3] = (int)(((tile_y >> 16) & 0xFFFFu)               // tensor_dim1[31:16]
          | ((tile_x & 0xFFFFu) << 16));                   // tile_dim0
    g1[4] = (int)(tile_y & 0xFFFFu);                       // tile_dim1 (tile_dim2=0)
    g1[5] = (int)(stride_elems & 0xFFFFFFFFu);             // tensor_dim0_stride lo
    g1[6] = (int)((stride_elems >> 32) & 0xFFFFu);         // stride hi (stride1=0)
    g1[7] = 0;

    i32x4 z4 = {};
#if defined(__clang_major__) && (__clang_major__ >= 23)
    i32x8 z8 = {};
    __builtin_amdgcn_tensor_load_to_lds(g0, g1, z4, z4, z8, 0);
#else
    __builtin_amdgcn_tensor_load_to_lds(g0, g1, z4, z4, 0);
#endif
}

// ---------------------------------------------------------------------------
// Weight transpose + f32 -> bf16 convert:  Wt[n*K + k] = W[k*N + n]
// ---------------------------------------------------------------------------
__global__ void transpose_bf16_kernel(const float* __restrict__ W,
                                      __bf16* __restrict__ Wt, int K, int N) {
    int n = blockIdx.x * 16 + threadIdx.x;
    int k = blockIdx.y * 16 + threadIdx.y;
    if (n < N && k < K)
        Wt[(size_t)n * K + k] = (__bf16)W[(size_t)k * N + n];
}

// ---------------------------------------------------------------------------
// LayerNorm (f32 in) -> bf16 out. One block per row of D_ elements.
// ---------------------------------------------------------------------------
__global__ void __launch_bounds__(256)
ln_bf16_kernel(const float* __restrict__ x, const float* __restrict__ g,
               const float* __restrict__ beta, __bf16* __restrict__ out) {
    __shared__ float red[256];
    const int row = blockIdx.x;
    const float* xr = x + (size_t)row * D_;

    float s = 0.f;
    for (int i = threadIdx.x; i < D_; i += 256) s += xr[i];
    red[threadIdx.x] = s; __syncthreads();
    for (int o = 128; o > 0; o >>= 1) {
        if (threadIdx.x < o) red[threadIdx.x] += red[threadIdx.x + o];
        __syncthreads();
    }
    const float mu = red[0] * (1.0f / D_);
    __syncthreads();

    float v = 0.f;
    for (int i = threadIdx.x; i < D_; i += 256) { float d = xr[i] - mu; v += d * d; }
    red[threadIdx.x] = v; __syncthreads();
    for (int o = 128; o > 0; o >>= 1) {
        if (threadIdx.x < o) red[threadIdx.x] += red[threadIdx.x + o];
        __syncthreads();
    }
    const float rstd = rsqrtf(red[0] * (1.0f / D_) + 1e-5f);

    for (int i = threadIdx.x; i < D_; i += 256)
        out[(size_t)row * D_ + i] = (__bf16)((xr[i] - mu) * rstd * g[i] + beta[i]);
}

// ---------------------------------------------------------------------------
// T5 relative position bias table: tab[h*4096 + (q-k)+2048] = rel_emb[bucket][h]
// ---------------------------------------------------------------------------
__global__ void relbias_kernel(const float* __restrict__ rel_emb,
                               float* __restrict__ tab) {
    int tid = blockIdx.x * 256 + threadIdx.x;
    if (tid >= H_ * 4096) return;
    int h   = tid >> 12;
    int n   = (tid & 4095) - 2048;
    int ret = (n < 0) ? 16 : 0;
    int na  = (n < 0) ? -n : n;
    int bucket;
    if (na < 8) {
        bucket = na;
    } else {
        float vl = 8.0f + __logf((float)na / 8.0f + 1e-6f) * (8.0f / __logf(16.0f));
        int bi = (int)vl;
        bucket = bi < 15 ? bi : 15;
    }
    tab[tid] = rel_emb[(bucket + ret) * H_ + h];
}

// ---------------------------------------------------------------------------
// WMMA bf16 GEMM:  C[M,N] = A[M,K](bf16) * Bt[N,K]^T(bf16) + bias (+GELU)(+res)
// Block: 256 threads = 8 waves, block tile 128x128, BK=32, double-buffered LDS.
// Wave (wm = w&3, wn = w>>2) computes 32x64 -> 8x v_wmma_f32_16x16x32_bf16/K-step.
// Tiles stream in via the Tensor Data Mover (one wave issues, TENSORcnt sync).
// ---------------------------------------------------------------------------
template <bool GELU, bool RES, bool OUTBF>
__global__ void __launch_bounds__(256)
gemm_wmma(const __bf16* __restrict__ A, const __bf16* __restrict__ Bt,
          const float* __restrict__ bias, const float* __restrict__ res,
          float* __restrict__ outF, __bf16* __restrict__ outB,
          int M, int N, int K) {
    __shared__ __bf16 As[2][128 * 32];
    __shared__ __bf16 Bs[2][128 * 32];

    const int t    = threadIdx.x;
    const int lane = t & 31, w = t >> 5;
    const int l16  = lane & 15, half = lane >> 4;
    const int wm   = w & 3, wn = w >> 2;
    const int m0   = blockIdx.y * 128, n0 = blockIdx.x * 128;
    const int nk   = K >> 5;

    v8f acc0[4] = {}, acc1[4] = {};

#if USE_TDM
    if (w == 0) {
        tdm_load_2d_bf16(&As[0][0], A  + (size_t)m0 * K, 32, 128, (unsigned long long)K);
        tdm_load_2d_bf16(&Bs[0][0], Bt + (size_t)n0 * K, 32, 128, (unsigned long long)K);
    }
#endif

    for (int kt = 0; kt < nk; ++kt) {
        const int buf = kt & 1;
#if USE_TDM
        if (w == 0) __builtin_amdgcn_s_wait_tensorcnt(0);
        __syncthreads();                       // tile kt visible to all waves
        if (w == 0 && kt + 1 < nk) {
            tdm_load_2d_bf16(&As[buf ^ 1][0], A  + (size_t)m0 * K + (kt + 1) * 32,
                             32, 128, (unsigned long long)K);
            tdm_load_2d_bf16(&Bs[buf ^ 1][0], Bt + (size_t)n0 * K + (kt + 1) * 32,
                             32, 128, (unsigned long long)K);
        }
#else
        __syncthreads();
#pragma unroll
        for (int i = 0; i < 2; ++i) {          // 4096 bf16 per tile, 8/thread/iter
            int e = (t + i * 256) * 8;
            int r = e >> 5, c = e & 31;
            *(uint4*)(&As[buf][r * 32 + c]) = *(const uint4*)(A  + (size_t)(m0 + r) * K + kt * 32 + c);
            *(uint4*)(&Bs[buf][r * 32 + c]) = *(const uint4*)(Bt + (size_t)(n0 + r) * K + kt * 32 + c);
        }
        __syncthreads();
#endif
        v16bf a0 = load_frag(&As[buf][(wm * 32 +      l16) * 32 + half * 8]);
        v16bf a1 = load_frag(&As[buf][(wm * 32 + 16 + l16) * 32 + half * 8]);
#pragma unroll
        for (int tn = 0; tn < 4; ++tn) {
            v16bf bfr = load_frag(&Bs[buf][(wn * 64 + tn * 16 + l16) * 32 + half * 8]);
            acc0[tn] = __builtin_amdgcn_wmma_f32_16x16x32_bf16(
                false, a0, false, bfr, (short)0, acc0[tn], false, false);
            acc1[tn] = __builtin_amdgcn_wmma_f32_16x16x32_bf16(
                false, a1, false, bfr, (short)0, acc1[tn], false, false);
        }
    }

#pragma unroll
    for (int sm = 0; sm < 2; ++sm) {
#pragma unroll
        for (int tn = 0; tn < 4; ++tn) {
#pragma unroll
            for (int r = 0; r < 8; ++r) {
                int row = m0 + wm * 32 + sm * 16 + r + 8 * half;
                int col = n0 + wn * 64 + tn * 16 + l16;
                float v = (sm ? acc1[tn][r] : acc0[tn][r]) + bias[col];
                if (GELU) v = 0.5f * v * (1.0f + erff(v * 0.70710678118654752f));
                if (RES)  v += res[(size_t)row * N + col];
                if (OUTBF) outB[(size_t)row * N + col] = (__bf16)v;
                else       outF[(size_t)row * N + col] = v;
            }
        }
    }
}

// ---------------------------------------------------------------------------
// Flash attention with WMMA. Block: (64 q rows, head h, batch b), 128 thr/4 waves.
// Wave owns 16 q rows. KV blocks of 32, double-buffered: TDM prefetches the
// next K tile and all threads stage the next V tile (transposed) while the
// current block computes. Rel-bias window + additive mask staged in LDS once.
// ---------------------------------------------------------------------------
__global__ void __launch_bounds__(128)
attn_wmma(const __bf16* __restrict__ Q, const __bf16* __restrict__ Kb,
          const __bf16* __restrict__ V, const float* __restrict__ relTab,
          const unsigned char* __restrict__ mask, __bf16* __restrict__ O) {
    __shared__ __bf16 Qs[64 * 64];        // [q][d]
    __shared__ __bf16 Ks[2][32 * 64];     // [kk][d]
    __shared__ __bf16 Vts[2][64 * 32];    // [d][kk]  (transposed on store)
    __shared__ __bf16 Ps[4][16 * 32];     // per-wave P tile [q][kk]
    __shared__ float  relW[2112];         // rel-bias window for this q block
    __shared__ float  maskW[2048];        // additive mask: 0 or -3e38

    const int t    = threadIdx.x;
    const int lane = t & 31, w = t >> 5;
    const int l16  = lane & 15, half = lane >> 4;
    const int q0 = blockIdx.x * 64, h = blockIdx.y, b = blockIdx.z;
    const size_t base = (size_t)b * S_ * D_ + (size_t)h * DH_;   // + s*D_ + d

    auto loadVt = [&](int k0, __bf16* dst) {
#pragma unroll
        for (int i = 0; i < 2; ++i) {
            int e = (t + i * 128) * 8;
            int r = e >> 6, c = e & 63;            // r = kk, c = d
            union { uint4 u; __bf16 e8[8]; } vv;
            vv.u = *(const uint4*)(V + base + (size_t)(k0 + r) * D_ + c);
#pragma unroll
            for (int j = 0; j < 8; ++j) dst[(c + j) * 32 + r] = vv.e8[j];
        }
    };

    // Prologue: TDM Q tile + K(0) tile; stage V(0), rel-bias window, mask row.
    if (w == 0) {
        tdm_load_2d_bf16(Qs, Q + base + (size_t)q0 * D_, 64, 64, (unsigned long long)D_);
        tdm_load_2d_bf16(Ks[0], Kb + base, 64, 32, (unsigned long long)D_);
    }
    const float* relh = relTab + h * 4096;
    for (int i = t; i < 2111; i += 128) relW[i] = relh[q0 + 1 + i];
    const unsigned char* maskb = mask + (size_t)b * S_;
    for (int i = t; i < 2048; i += 128) maskW[i] = maskb[i] ? 0.f : -3.0e38f;
    loadVt(0, Vts[0]);
    if (w == 0) __builtin_amdgcn_s_wait_tensorcnt(0);
    __syncthreads();

    float m_r[8], l_r[8];
    v8f acc_o[4] = {};
#pragma unroll
    for (int r = 0; r < 8; ++r) { m_r[r] = -3.0e38f; l_r[r] = 0.f; }

    const int nkt = S_ / 32;
    for (int kt = 0; kt < nkt; ++kt) {
        const int buf = kt & 1;
        const int k0  = kt * 32;

        // Prefetch next tiles into the other buffer while we compute this one.
        if (kt + 1 < nkt) {
            if (w == 0)
                tdm_load_2d_bf16(Ks[buf ^ 1], Kb + base + (size_t)(k0 + 32) * D_,
                                 64, 32, (unsigned long long)D_);
            loadVt(k0 + 32, Vts[buf ^ 1]);
        }

        // S = Q * K^T : A = Q rows (16x64), B col n = kk -> Ks row, K-dim = d
        v8f sacc[2] = {};
#pragma unroll
        for (int dk = 0; dk < 64; dk += 32) {
            v16bf af = load_frag(Qs + (w * 16 + l16) * 64 + dk + half * 8);
#pragma unroll
            for (int tn = 0; tn < 2; ++tn) {
                v16bf bfr = load_frag(&Ks[buf][(tn * 16 + l16) * 64 + dk + half * 8]);
                sacc[tn] = __builtin_amdgcn_wmma_f32_16x16x32_bf16(
                    false, af, false, bfr, (short)0, sacc[tn], false, false);
            }
        }

        // bias + mask + online softmax (row = r + 8*half, col = tn*16 + l16)
        float madd[2];
#pragma unroll
        for (int tn = 0; tn < 2; ++tn) madd[tn] = maskW[k0 + tn * 16 + l16];

        float p[2][8];
#pragma unroll
        for (int r = 0; r < 8; ++r) {
            int wq = w * 16 + r + 8 * half;       // q row within block
#pragma unroll
            for (int tn = 0; tn < 2; ++tn) {
                int kpos = k0 + tn * 16 + l16;
                p[tn][r] = sacc[tn][r] * 0.125f + relW[wq - kpos + 2047] + madd[tn];
            }
            float mx = fmaxf(p[0][r], p[1][r]);
#pragma unroll
            for (int off = 1; off < 16; off <<= 1)
                mx = fmaxf(mx, __shfl_xor(mx, off, 32));
            float mnew  = fmaxf(m_r[r], mx);
            float scale = __expf(m_r[r] - mnew);
            float sum = 0.f;
#pragma unroll
            for (int tn = 0; tn < 2; ++tn) {
                float e = __expf(p[tn][r] - mnew);
                p[tn][r] = e; sum += e;
            }
#pragma unroll
            for (int off = 1; off < 16; off <<= 1)
                sum += __shfl_xor(sum, off, 32);
            l_r[r] = l_r[r] * scale + sum;
            m_r[r] = mnew;
#pragma unroll
            for (int t4 = 0; t4 < 4; ++t4) acc_o[t4][r] *= scale;
        }

        // Stage P (per-wave region; DS ops of a wave are in-order -> no barrier)
#pragma unroll
        for (int r = 0; r < 8; ++r)
#pragma unroll
            for (int tn = 0; tn < 2; ++tn)
                Ps[w][(r + 8 * half) * 32 + tn * 16 + l16] = (__bf16)p[tn][r];

        // O += P * V : A = P (16x32), B col n = d -> Vts row, K-dim = kk
        v16bf pf = load_frag(Ps[w] + l16 * 32 + half * 8);
#pragma unroll
        for (int t4 = 0; t4 < 4; ++t4) {
            v16bf vf = load_frag(&Vts[buf][(t4 * 16 + l16) * 32 + half * 8]);
            acc_o[t4] = __builtin_amdgcn_wmma_f32_16x16x32_bf16(
                false, pf, false, vf, (short)0, acc_o[t4], false, false);
        }

        if (w == 0 && kt + 1 < nkt) __builtin_amdgcn_s_wait_tensorcnt(0);
        __syncthreads();   // everyone done with buf; next tiles complete
    }

#pragma unroll
    for (int r = 0; r < 8; ++r) {
        int qpos = q0 + w * 16 + r + 8 * half;
        float inv = 1.0f / l_r[r];
#pragma unroll
        for (int t4 = 0; t4 < 4; ++t4) {
            int d = t4 * 16 + l16;
            O[base + (size_t)qpos * D_ + d] = (__bf16)(acc_o[t4][r] * inv);
        }
    }
}

// ---------------------------------------------------------------------------
extern "C" void kernel_launch(void* const* d_in, const int* in_sizes, int n_in,
                              void* d_out, int out_size, void* d_ws, size_t ws_size,
                              hipStream_t stream) {
    (void)in_sizes; (void)n_in; (void)out_size; (void)ws_size;
    const float* x     = (const float*)d_in[0];
    const float* Wq    = (const float*)d_in[1];
    const float* bq    = (const float*)d_in[2];
    const float* Wk    = (const float*)d_in[3];
    const float* bk    = (const float*)d_in[4];
    const float* Wv    = (const float*)d_in[5];
    const float* bv    = (const float*)d_in[6];
    const float* Wo    = (const float*)d_in[7];
    const float* bo    = (const float*)d_in[8];
    const float* W1    = (const float*)d_in[9];
    const float* b1    = (const float*)d_in[10];
    const float* W2    = (const float*)d_in[11];
    const float* b2    = (const float*)d_in[12];
    const float* g1    = (const float*)d_in[13];
    const float* beta1 = (const float*)d_in[14];
    const float* g2    = (const float*)d_in[15];
    const float* beta2 = (const float*)d_in[16];
    const float* rel   = (const float*)d_in[17];
    const unsigned char* pmask = (const unsigned char*)d_in[18];
    float* out = (float*)d_out;

    char* ws = (char*)d_ws;
    size_t off = 0;
    auto alloc = [&](size_t bytes) -> void* {
        void* p = ws + off;
        off = (off + bytes + 255) & ~(size_t)255;
        return p;
    };

    __bf16* WqT = (__bf16*)alloc((size_t)D_ * D_ * 2);
    __bf16* WkT = (__bf16*)alloc((size_t)D_ * D_ * 2);
    __bf16* WvT = (__bf16*)alloc((size_t)D_ * D_ * 2);
    __bf16* WoT = (__bf16*)alloc((size_t)D_ * D_ * 2);
    __bf16* W1T = (__bf16*)alloc((size_t)D_ * FF_ * 2);
    __bf16* W2T = (__bf16*)alloc((size_t)FF_ * D_ * 2);
    float*  relTab = (float*)alloc((size_t)H_ * 4096 * 4);
    float*  x1  = (float*)alloc((size_t)M_ * D_ * 4);
    __bf16* hln = (__bf16*)alloc((size_t)M_ * D_ * 2);       // h1 then h2
    __bf16* qb  = (__bf16*)alloc((size_t)M_ * D_ * 2);
    __bf16* kb  = (__bf16*)alloc((size_t)M_ * D_ * 2);
    __bf16* vb  = (__bf16*)alloc((size_t)M_ * D_ * 2);
    __bf16* ao  = (__bf16*)alloc((size_t)M_ * D_ * 2);
    __bf16* ffa = qb;   // 32MB FFN activation aliases dead q/k/v/ao region

    dim3 tb(16, 16);
    transpose_bf16_kernel<<<dim3(D_ / 16, D_ / 16), tb, 0, stream>>>(Wq, WqT, D_, D_);
    transpose_bf16_kernel<<<dim3(D_ / 16, D_ / 16), tb, 0, stream>>>(Wk, WkT, D_, D_);
    transpose_bf16_kernel<<<dim3(D_ / 16, D_ / 16), tb, 0, stream>>>(Wv, WvT, D_, D_);
    transpose_bf16_kernel<<<dim3(D_ / 16, D_ / 16), tb, 0, stream>>>(Wo, WoT, D_, D_);
    transpose_bf16_kernel<<<dim3(FF_ / 16, D_ / 16), tb, 0, stream>>>(W1, W1T, D_, FF_);
    transpose_bf16_kernel<<<dim3(D_ / 16, FF_ / 16), tb, 0, stream>>>(W2, W2T, FF_, D_);

    relbias_kernel<<<(H_ * 4096 + 255) / 256, 256, 0, stream>>>(rel, relTab);

    // h1 = LN(x)
    ln_bf16_kernel<<<M_, 256, 0, stream>>>(x, g1, beta1, hln);

    // Q/K/V projections (bf16 out)
    gemm_wmma<false, false, true><<<dim3(D_ / 128, M_ / 128), 256, 0, stream>>>(
        hln, WqT, bq, nullptr, nullptr, qb, M_, D_, D_);
    gemm_wmma<false, false, true><<<dim3(D_ / 128, M_ / 128), 256, 0, stream>>>(
        hln, WkT, bk, nullptr, nullptr, kb, M_, D_, D_);
    gemm_wmma<false, false, true><<<dim3(D_ / 128, M_ / 128), 256, 0, stream>>>(
        hln, WvT, bv, nullptr, nullptr, vb, M_, D_, D_);

    // flash attention
    attn_wmma<<<dim3(S_ / 64, H_, B_), 128, 0, stream>>>(qb, kb, vb, relTab, pmask, ao);

    // x1 = x + ao @ Wo + bo   (f32 out)
    gemm_wmma<false, true, false><<<dim3(D_ / 128, M_ / 128), 256, 0, stream>>>(
        ao, WoT, bo, x, x1, nullptr, M_, D_, D_);

    // h2 = LN(x1)
    ln_bf16_kernel<<<M_, 256, 0, stream>>>(x1, g2, beta2, hln);

    // ffa = gelu(h2 @ W1 + b1)   (bf16 out)
    gemm_wmma<true, false, true><<<dim3(FF_ / 128, M_ / 128), 256, 0, stream>>>(
        hln, W1T, b1, nullptr, nullptr, ffa, M_, FF_, D_);

    // out = x1 + ffa @ W2 + b2   (f32 out)
    gemm_wmma<false, true, false><<<dim3(D_ / 128, M_ / 128), 256, 0, stream>>>(
        ffa, W2T, b2, x1, out, nullptr, M_, D_, FF_);
}
